// NTXentLoss_21758304321928
// MI455X (gfx1250) — matrix operand order repaired and checked by reading, not compiled, not run on hardware
//
#include <hip/hip_runtime.h>
#include <hip/hip_bf16.h>

// NT-Xent loss, fused: never materializes the 8192x8192 sim matrix.
// Phase 1: row-normalize [8192,256] f32 -> f16 reps (4 MB, L2-resident).
// Phase 1b: positives  pos[m] = dot(z[m], z[m^N])  (tiny).
// Phase 2: tiled Z*Z^T via V_WMMA_F32_16X16X32_F16 with fused exp/mask/rowsum,
//          double-buffered LDS B tiles, one barrier per tile.
// Phase 3: scalar reduction  loss = mean( log(denom) - pos/T ),  T = 0.5.

typedef __attribute__((ext_vector_type(16))) _Float16 v16h;
typedef __attribute__((ext_vector_type(8)))  _Float16 v8h;
typedef __attribute__((ext_vector_type(8)))  float    v8f;
typedef __attribute__((ext_vector_type(4)))  unsigned int v4u;

#define N_BATCH   4096
#define N2        8192           // 2*N
#define DDIM      256            // embedding dim
#define INV_T     2.0f           // 1 / 0.5
#define BSTRIDE   264            // halves per B-tile column in LDS (256 + 8 pad, keeps 16B align)
#define COL_SPLITS 8             // column space split across blockIdx.y
#define ROWS_PER_WG 128          // 8 waves * 16 rows

// ---------------------------------------------------------------------------
// Kernel 1: L2-normalize each row, emit f16, zero denom accumulator.
// One wave (32 lanes) per row; each lane owns 8 consecutive floats.
// ---------------------------------------------------------------------------
__global__ __launch_bounds__(256)
void ntxent_norm_kernel(const float* __restrict__ emb_i,
                        const float* __restrict__ emb_j,
                        _Float16* __restrict__ reps,
                        float* __restrict__ denom) {
  const int wave = threadIdx.x >> 5;
  const int lane = threadIdx.x & 31;
  const int row  = blockIdx.x * 8 + wave;           // 1024 blocks * 8 waves = 8192 rows

  const float* __restrict__ src =
      (row < N_BATCH) ? (emb_i + (size_t)row * DDIM)
                      : (emb_j + (size_t)(row - N_BATCH) * DDIM);

  const float4 x0 = *(const float4*)(src + lane * 8);
  const float4 x1 = *(const float4*)(src + lane * 8 + 4);

  float ss = x0.x * x0.x + x0.y * x0.y + x0.z * x0.z + x0.w * x0.w +
             x1.x * x1.x + x1.y * x1.y + x1.z * x1.z + x1.w * x1.w;
  #pragma unroll
  for (int off = 16; off >= 1; off >>= 1)
    ss += __shfl_xor(ss, off, 32);

  const float scale = 1.0f / fmaxf(sqrtf(ss), 1e-12f);

  v8h h;
  h[0] = (_Float16)(x0.x * scale);  h[1] = (_Float16)(x0.y * scale);
  h[2] = (_Float16)(x0.z * scale);  h[3] = (_Float16)(x0.w * scale);
  h[4] = (_Float16)(x1.x * scale);  h[5] = (_Float16)(x1.y * scale);
  h[6] = (_Float16)(x1.z * scale);  h[7] = (_Float16)(x1.w * scale);
  *(v8h*)(reps + (size_t)row * DDIM + lane * 8) = h;

  if (lane == 0) denom[row] = 0.0f;
}

// ---------------------------------------------------------------------------
// Kernel 1b: positives.  pos[m] = dot(z[m], z[m ^ N]).  One wave per row.
// (Removed from the GEMM hot loop: the per-tile divergent store was costing
//  8 exec-mask round-trips per iteration.)
// ---------------------------------------------------------------------------
__global__ __launch_bounds__(256)
void ntxent_pos_kernel(const _Float16* __restrict__ reps,
                       float* __restrict__ pos) {
  const int wave = threadIdx.x >> 5;
  const int lane = threadIdx.x & 31;
  const int m = blockIdx.x * 8 + wave;

  const v8h a = *(const v8h*)(reps + (size_t)m * DDIM + lane * 8);
  const v8h b = *(const v8h*)(reps + (size_t)(m ^ N_BATCH) * DDIM + lane * 8);

  float s = 0.0f;
  #pragma unroll
  for (int i = 0; i < 8; ++i)
    s += (float)a[i] * (float)b[i];

  #pragma unroll
  for (int off = 16; off >= 1; off >>= 1)
    s += __shfl_xor(s, off, 32);

  if (lane == 0) pos[m] = s;
}

// ---------------------------------------------------------------------------
// Kernel 2: fused  sim = reps @ reps^T  ->  exp(sim/T), mask diag, row-sum.
// Block = 256 threads = 8 waves. Each wave owns a 16-row strip (A frags held
// in 64 VGPRs for the whole kernel). All waves share each 16-column B tile,
// staged through double-buffered LDS (one barrier per tile). blockIdx.y
// splits the 8192 columns 8 ways; partials combine with global f32 atomics.
//
// CDNA5 f16 operand lane layouts (CDNA5 ISA):
//   A 16x32: lane<16 -> row M=lane,    K = k0+{0..7}, k0+{16..23}
//            lane>=16-> row M=lane-16, K = k0+{8..15}, k0+{24..31}
//   B 32x16 (kernel's own K order, A same): lane<16 -> col N=lane,    K = k0+{0..15}   (contiguous 32B)
//            lane>=16-> col N=lane-16, K = k0+{16..31} (used by this kernel, not the ISA table)
//   C/D 16x16 f32: VGPR r, lane<16 -> (M=r,   N=lane)
//                          lane>=16-> (M=r+8, N=lane-16)
// ---------------------------------------------------------------------------
__global__ __launch_bounds__(256, 1)
void ntxent_sim_kernel(const _Float16* __restrict__ reps,
                       float* __restrict__ denom) {
  __shared__ _Float16 bsh[2][16 * BSTRIDE];         // 2 x 8448 B B-tile stages

  const int tid   = threadIdx.x;
  const int wave  = tid >> 5;
  const int lane  = tid & 31;
  const int lhalf = lane >> 4;                      // 0: lanes 0-15, 1: lanes 16-31
  const int l16   = lane & 15;

  const int mBase = blockIdx.x * ROWS_PER_WG + wave * 16;

  // ---- preload A fragments: this lane's row, all K=256 (8 frags of K=32) ----
  v16h a[8];
  {
    const _Float16* __restrict__ arow =
        reps + (size_t)(mBase + l16) * DDIM + lhalf * 16;
    #pragma unroll
    for (int kf = 0; kf < 8; ++kf) {
      const v8h lo = *(const v8h*)(arow + kf * 32);        // K k0+{0..7}  (+16 upper half)
      const v8h hi = *(const v8h*)(arow + kf * 32 + 8);    // K k0+{8..15} (+16 upper half)
      v16h f;
      #pragma unroll
      for (int i = 0; i < 8; ++i) { f[i] = lo[i]; f[i + 8] = hi[i]; }
      a[kf] = f;
    }
  }

  float acc[8];
  #pragma unroll
  for (int r = 0; r < 8; ++r) acc[r] = 0.0f;

  const int tilesPerSplit = (N2 / COL_SPLITS) / 16;   // 64
  const int tileBeg = blockIdx.y * tilesPerSplit;

  // cooperative B-tile copy coordinates: 32B per thread
  const int ccol = tid >> 4;
  const int cko  = (tid & 15) * 16;

  // ---- prologue: stage tile 0 into buffer 0 ----
  {
    const v4u* __restrict__ src =
        (const v4u*)(reps + (size_t)(tileBeg * 16 + ccol) * DDIM + cko);
    v4u* dst = (v4u*)(&bsh[0][ccol * BSTRIDE + cko]);
    dst[0] = src[0];
    dst[1] = src[1];
  }
  __syncthreads();

  for (int t = 0; t < tilesPerSplit; ++t) {
    const int cur = t & 1;
    const int nBase = (tileBeg + t) * 16;

    // ---- prefetch tile t+1 from global into registers (hides L2 latency) ----
    v4u p0, p1;
    const bool haveNext = (t + 1) < tilesPerSplit;
    if (haveNext) {
      const v4u* __restrict__ src =
          (const v4u*)(reps + (size_t)((nBase + 16) + ccol) * DDIM + cko);
      p0 = src[0];
      p1 = src[1];
    }

    // ---- 16x16 tile of sim via 8 chained WMMAs over K=256 ----
    v8f c = {};
    const _Float16* bcol = &bsh[cur][l16 * BSTRIDE + lhalf * 16];
    #pragma unroll
    for (int kf = 0; kf < 8; ++kf) {
      const v16h b = *(const v16h*)(bcol + kf * 32);
      c = __builtin_amdgcn_wmma_f32_16x16x32_f16(
              /*neg_a=*/false, a[kf], /*neg_b=*/false, b,
              /*c_mod=*/(short)0, c, /*reuse_a=*/false, /*reuse_b=*/false);
    }

    // ---- fused epilogue: exp(s/T), branchless diagonal mask ----
    const int n = nBase + l16;
    #pragma unroll
    for (int r = 0; r < 8; ++r) {
      const int mr = mBase + r + lhalf * 8;
      const float e = __expf(c[r] * INV_T);
      acc[r] += (n == mr) ? 0.0f : e;               // 1 - eye mask (v_cndmask)
    }

    // ---- stage tile t+1 into the other buffer; single barrier per tile ----
    if (haveNext) {
      v4u* dst = (v4u*)(&bsh[cur ^ 1][ccol * BSTRIDE + cko]);
      dst[0] = p0;
      dst[1] = p1;
    }
    __syncthreads();
  }

  // ---- reduce row partial sums across the 16 lanes of each half ----
  #pragma unroll
  for (int r = 0; r < 8; ++r) {
    float v = acc[r];
    #pragma unroll
    for (int off = 8; off >= 1; off >>= 1)
      v += __shfl_xor(v, off, 32);                  // stays inside 16-lane half
    if (l16 == 0)
      atomicAdd(&denom[mBase + r + lhalf * 8], v);  // combine column splits
  }
}

// ---------------------------------------------------------------------------
// Kernel 3: loss = (1/2N) * sum_m [ log(denom[m]) - pos[m]/T ]
// ---------------------------------------------------------------------------
__global__ __launch_bounds__(256)
void ntxent_loss_kernel(const float* __restrict__ denom,
                        const float* __restrict__ pos,
                        float* __restrict__ out) {
  __shared__ float red[8];
  const int wave = threadIdx.x >> 5;
  const int lane = threadIdx.x & 31;

  float s = 0.0f;
  for (int m = threadIdx.x; m < N2; m += 256)
    s += logf(denom[m]) - pos[m] * INV_T;

  #pragma unroll
  for (int off = 16; off >= 1; off >>= 1)
    s += __shfl_xor(s, off, 32);
  if (lane == 0) red[wave] = s;
  __syncthreads();

  if (threadIdx.x == 0) {
    float tot = 0.0f;
    #pragma unroll
    for (int w = 0; w < 8; ++w) tot += red[w];
    out[0] = tot / (float)N2;
  }
}

// ---------------------------------------------------------------------------
extern "C" void kernel_launch(void* const* d_in, const int* in_sizes, int n_in,
                              void* d_out, int out_size, void* d_ws, size_t ws_size,
                              hipStream_t stream) {
  (void)in_sizes; (void)n_in; (void)out_size; (void)ws_size;

  const float* emb_i = (const float*)d_in[0];   // [4096, 256] f32
  const float* emb_j = (const float*)d_in[1];   // [4096, 256] f32
  float* out = (float*)d_out;                   // scalar f32

  // workspace layout: f16 reps (4 MB) | denom[8192] f32 | pos[8192] f32
  _Float16* reps = (_Float16*)d_ws;
  float* denom   = (float*)((char*)d_ws + (size_t)N2 * DDIM * sizeof(_Float16));
  float* pos     = denom + N2;

  ntxent_norm_kernel<<<dim3(N2 / 8), dim3(256), 0, stream>>>(emb_i, emb_j, reps, denom);

  ntxent_pos_kernel<<<dim3(N2 / 8), dim3(256), 0, stream>>>(reps, pos);

  ntxent_sim_kernel<<<dim3(N2 / ROWS_PER_WG, COL_SPLITS), dim3(256), 0, stream>>>(
      reps, denom);

  ntxent_loss_kernel<<<dim3(1), dim3(256), 0, stream>>>(denom, pos, out);
}
